// DecoderOutputLayer_36335423324309
// MI455X (gfx1250) — matrix operand order repaired and checked by reading, not compiled, not run on hardware
//
#include <hip/hip_runtime.h>
#include <hip/hip_bf16.h>
#include <math.h>

// ---------------------------------------------------------------------------
// Fused attention-pooling head for MI455X (gfx1250, wave32, WMMA + TDM).
//   scores = U @ x^T          [B,Y,L] (never materialized)
//   F      = final_w @ x^T    [B,Y,L] (never materialized)
//   y[b,y] = softmax_l(scores) . F + bias  (online softmax over L tiles)
//   loss   = mean BCEWithLogits(y, target)
//
// Fast path: one-time f32->bf16 conversion into d_ws, then the main kernel
// stages bf16 tiles into double-buffered LDS with TENSOR_LOAD_TO_LDS (TDM),
// overlapping DMA with the v_wmma_f32_16x16x32_bf16 stream.
// Fallback path (small d_ws): convert-in-kernel staging.
// ---------------------------------------------------------------------------

typedef __attribute__((ext_vector_type(16))) __bf16 v16bf;
typedef __attribute__((ext_vector_type(8)))  __bf16 v8bf;
typedef __attribute__((ext_vector_type(4)))  __bf16 v4bf;
typedef __attribute__((ext_vector_type(8)))  float  v8f;
typedef __attribute__((ext_vector_type(4)))  float  v4f;
typedef __attribute__((ext_vector_type(4)))  unsigned v4u;
typedef __attribute__((ext_vector_type(8)))  int    v8i;
typedef __attribute__((ext_vector_type(4)))  int    v4i;

#define B_   16
#define L_   2048
#define D_   512
#define Y_   8921
#define YBLK 128      // Y rows per workgroup (8 waves x 16)
#define LT   128      // L columns per tile (8 wmma col-tiles)
#define KC   64       // K (=D) chunk staged in LDS per pass
#define PITCH 72      // bf16 elements per LDS row (64 + 8 pad -> 144B rows)
#define TILE (YBLK * PITCH)          // elems per staged tile (9216)
#define NPH  ((L_ / LT) * (D_ / KC)) // 16 * 8 = 128 phases

__device__ __forceinline__ __bf16 f2bf(float f) {
    union { float f; unsigned u; } v; v.f = f;
    unsigned r = (v.u + 0x7FFFu + ((v.u >> 16) & 1u)) >> 16;  // RNE
    union { unsigned short s; __bf16 b; } o; o.s = (unsigned short)r;
    return o.b;
}

// ---------------------------------------------------------------------------
// TDM: issue a 2D bf16 tile load Global->LDS with row padding.
//   tile = tile_d1 rows x tile_d0 elems, row stride in memory = stride0 elems.
//   LDS: rows packed at 128B + 16B pad = 144B pitch (matches PITCH=72 bf16).
//   Rows >= tensor_d1 read as zero (handles the Y edge).
// ---------------------------------------------------------------------------
__device__ __forceinline__ void tdm_load_tile(unsigned lds_byte_off,
                                              const __bf16* gsrc,
                                              unsigned tensor_d0, unsigned tensor_d1,
                                              unsigned tile_d0,   unsigned tile_d1,
                                              unsigned stride0) {
    unsigned long long ga = (unsigned long long)(size_t)gsrc;
    v4u g0;
    g0[0] = 1u;                                             // count=1, user D#
    g0[1] = lds_byte_off;                                   // lds_addr
    g0[2] = (unsigned)(ga & 0xFFFFFFFFu);                   // global_addr lo
    g0[3] = ((unsigned)(ga >> 32) & 0x01FFFFFFu) | 0x80000000u; // hi + type=2
    v8i g1;
    // data_size=1 (2B) | pad_enable | pad_interval=4 (32 DW=128B rows)
    // | pad_amount=3 (4 DW = 16B pad)
    g1[0] = (1 << 16) | (1 << 20) | (4 << 22) | (3 << 25);
    g1[1] = (int)((tensor_d0 & 0xFFFFu) << 16);             // td0[15:0]
    g1[2] = (int)((tensor_d0 >> 16) | ((tensor_d1 & 0xFFFFu) << 16));
    g1[3] = (int)((tensor_d1 >> 16) | (tile_d0 << 16));     // td1 hi | tile_d0
    g1[4] = (int)(tile_d1 & 0xFFFFu);                       // tile_d1, tile_d2=0
    g1[5] = (int)stride0;                                   // dim0 stride lo32
    g1[6] = 0;                                              // stride hi / dim1 stride
    g1[7] = 0;
    v4i g2 = {0, 0, 0, 0};
    v4i g3 = {0, 0, 0, 0};
    v8i g4 = {0, 0, 0, 0, 0, 0, 0, 0};   // unused extra group (clang-23 form)
    __builtin_amdgcn_tensor_load_to_lds(g0, g1, g2, g3, g4, 0);
}

// ---------------------------------------------------------------------------
// Fast-path main kernel: bf16 inputs from d_ws, TDM double-buffered staging.
// ---------------------------------------------------------------------------
__global__ __launch_bounds__(256)
void attn_pool_wmma_tdm_kernel(const __bf16* __restrict__ xbf,
                               const __bf16* __restrict__ Ubf,
                               const __bf16* __restrict__ Fbf,
                               const float*  __restrict__ bias,
                               float*        __restrict__ yout) {
    extern __shared__ __bf16 lds[];    // [2 bufs][3 tiles][TILE]

    const int b    = blockIdx.y;
    const int y0   = blockIdx.x * YBLK;
    const int tid  = threadIdx.x;
    const int lane = tid & 31;
    const int w    = tid >> 5;
    const int half = lane >> 4;
    const int l16  = lane & 15;

    const unsigned lds_base = (unsigned)(size_t)&lds[0];    // LDS byte offset
    const unsigned rowsLeft = (unsigned)(Y_ - y0);          // rows avail (<=Y)

    float run_m[8], run_s[8], run_a[8];
    #pragma unroll
    for (int r = 0; r < 8; ++r) { run_m[r] = -1e30f; run_s[r] = 0.f; run_a[r] = 0.f; }

    const v8f vzero = {0.f,0.f,0.f,0.f,0.f,0.f,0.f,0.f};
    v8f accS[8], accF[8];

    // Issue all three tile DMAs for one phase into buffer `buf`.
    auto issue_phase = [&](int p, int buf) {
        const int l0 = (p >> 3) * LT;
        const int k0 = (p & 7) * KC;
        const unsigned base = lds_base + (unsigned)buf * (3u * TILE * 2u);
        // x tile: [LT rows][KC k], k contiguous in memory
        tdm_load_tile(base,                xbf + ((size_t)b * L_ + l0) * D_ + k0,
                      (unsigned)KC, (unsigned)(L_ - l0), KC, LT, (unsigned)D_);
        // U tile: [YBLK rows][KC k]; rows beyond Y read as zero
        tdm_load_tile(base + TILE * 2u,    Ubf + (size_t)y0 * D_ + k0,
                      (unsigned)KC, rowsLeft, KC, YBLK, (unsigned)D_);
        // final_w tile
        tdm_load_tile(base + TILE * 4u,    Fbf + (size_t)y0 * D_ + k0,
                      (unsigned)KC, rowsLeft, KC, YBLK, (unsigned)D_);
    };

    if (w == 0) issue_phase(0, 0);

    for (int p = 0; p < NPH; ++p) {
        if ((p & 7) == 0) {
            #pragma unroll
            for (int t = 0; t < 8; ++t) { accS[t] = vzero; accF[t] = vzero; }
        }
        if (w == 0) __builtin_amdgcn_s_wait_tensorcnt(0);   // this buffer ready
        __syncthreads();              // publish readiness; other buffer now idle
        if (w == 0 && p + 1 < NPH) issue_phase(p + 1, (p + 1) & 1);

        const __bf16* bx = lds + (p & 1) * 3 * TILE;
        const __bf16* bu = bx + TILE;
        const __bf16* bf = bx + 2 * TILE;

        #pragma unroll
        for (int ks = 0; ks < KC; ks += 32) {
            // A frag (16x32 bf16): lanes 0-15 K{0..7,16..23}, 16-31 K{8..15,24..31}
            const int arow = w * 16 + l16;
            const int aoff = ks + (half ? 8 : 0);
            v8bf alo = *(const v8bf*)&bu[arow * PITCH + aoff];
            v8bf ahi = *(const v8bf*)&bu[arow * PITCH + aoff + 16];
            v8bf flo = *(const v8bf*)&bf[arow * PITCH + aoff];
            v8bf fhi = *(const v8bf*)&bf[arow * PITCH + aoff + 16];
            v16bf aU, aF;
            #pragma unroll
            for (int i = 0; i < 8; ++i) {
                aU[i] = alo[i]; aU[8+i] = ahi[i];
                aF[i] = flo[i]; aF[8+i] = fhi[i];
            }
            // B frag (32x16 bf16): lanes 0-15 K{0..7,16..23}, 16-31 K{8..15,24..31}
            const int boff = ks + (half ? 8 : 0);
            #pragma unroll
            for (int lt = 0; lt < 8; ++lt) {
                const int brow = lt * 16 + l16;
                v8bf blo = *(const v8bf*)&bx[brow * PITCH + boff];
                v8bf bhi = *(const v8bf*)&bx[brow * PITCH + boff + 16];
                v16bf bb;
                #pragma unroll
                for (int i = 0; i < 8; ++i) { bb[i] = blo[i]; bb[8+i] = bhi[i]; }
                accS[lt] = __builtin_amdgcn_wmma_f32_16x16x32_bf16(
                    false, aU, false, bb, (short)0, accS[lt], false, false);
                accF[lt] = __builtin_amdgcn_wmma_f32_16x16x32_bf16(
                    false, aF, false, bb, (short)0, accF[lt], false, false);
            }
        }

        if ((p & 7) == 7) {   // L-tile complete -> online softmax update
            #pragma unroll
            for (int r = 0; r < 8; ++r) {
                float mx = -1e30f;
                #pragma unroll
                for (int t = 0; t < 8; ++t) mx = fmaxf(mx, accS[t][r]);
                #pragma unroll
                for (int dlt = 1; dlt < 16; dlt <<= 1)
                    mx = fmaxf(mx, __shfl_xor(mx, dlt, 32));
                float nm   = fmaxf(run_m[r], mx);
                float corr = __expf(run_m[r] - nm);
                float ps = 0.f, pa = 0.f;
                #pragma unroll
                for (int t = 0; t < 8; ++t) {
                    float e = __expf(accS[t][r] - nm);
                    ps += e;
                    pa += e * accF[t][r];
                }
                #pragma unroll
                for (int dlt = 1; dlt < 16; dlt <<= 1) {
                    ps += __shfl_xor(ps, dlt, 32);
                    pa += __shfl_xor(pa, dlt, 32);
                }
                run_s[r] = run_s[r] * corr + ps;
                run_a[r] = run_a[r] * corr + pa;
                run_m[r] = nm;
            }
        }
    }

    if (l16 == 0) {
        #pragma unroll
        for (int r = 0; r < 8; ++r) {
            int yg = y0 + w * 16 + r + half * 8;
            if (yg < Y_)
                yout[(size_t)b * Y_ + yg] = run_a[r] / run_s[r] + bias[yg];
        }
    }
}

// ---------------------------------------------------------------------------
// One-time f32 -> bf16 pack (fast path preprocessing).
// ---------------------------------------------------------------------------
__global__ __launch_bounds__(256)
void f32_to_bf16_kernel(const float* __restrict__ src,
                        __bf16* __restrict__ dst, int n4) {
    int i = blockIdx.x * 256 + threadIdx.x;
    const int stride = gridDim.x * 256;
    for (; i < n4; i += stride) {
        v4f v = *(const v4f*)(src + 4 * (size_t)i);
        v4bf o;
        o[0] = f2bf(v.x); o[1] = f2bf(v.y); o[2] = f2bf(v.z); o[3] = f2bf(v.w);
        *(v4bf*)(dst + 4 * (size_t)i) = o;
    }
}

// ---------------------------------------------------------------------------
// Fallback main kernel: convert-in-kernel LDS staging (no workspace needed).
// ---------------------------------------------------------------------------
__global__ __launch_bounds__(256)
void attn_pool_wmma_fallback_kernel(const float* __restrict__ x,
                                    const float* __restrict__ Uw,
                                    const float* __restrict__ Fw,
                                    const float* __restrict__ bias,
                                    float* __restrict__ yout) {
    __shared__ __bf16 ldsX[LT   * PITCH];
    __shared__ __bf16 ldsU[YBLK * PITCH];
    __shared__ __bf16 ldsF[YBLK * PITCH];

    const int b    = blockIdx.y;
    const int y0   = blockIdx.x * YBLK;
    const int tid  = threadIdx.x;
    const int lane = tid & 31;
    const int w    = tid >> 5;
    const int half = lane >> 4;
    const int l16  = lane & 15;

    float run_m[8], run_s[8], run_a[8];
    #pragma unroll
    for (int r = 0; r < 8; ++r) { run_m[r] = -1e30f; run_s[r] = 0.f; run_a[r] = 0.f; }

    const v8f vzero = {0.f,0.f,0.f,0.f,0.f,0.f,0.f,0.f};

    for (int l0 = 0; l0 < L_; l0 += LT) {
        v8f accS[8], accF[8];
        #pragma unroll
        for (int t = 0; t < 8; ++t) { accS[t] = vzero; accF[t] = vzero; }

        for (int k0 = 0; k0 < D_; k0 += KC) {
            __syncthreads();
            #pragma unroll
            for (int it = 0; it < 8; ++it) {
                int task = tid + it * 256;
                int row  = task >> 4;
                int g    = (task & 15) << 2;
                v4f v = *(const v4f*)(x + ((size_t)b * L_ + (l0 + row)) * D_ + k0 + g);
                __bf16* dst = &ldsX[row * PITCH + g];
                dst[0] = f2bf(v.x); dst[1] = f2bf(v.y);
                dst[2] = f2bf(v.z); dst[3] = f2bf(v.w);
            }
            #pragma unroll
            for (int it = 0; it < 8; ++it) {
                int task = tid + it * 256;
                int row  = task >> 4;
                int g    = (task & 15) << 2;
                int yg   = y0 + row;
                v4f vu = {0.f,0.f,0.f,0.f}, vf = {0.f,0.f,0.f,0.f};
                if (yg < Y_) {
                    vu = *(const v4f*)(Uw + (size_t)yg * D_ + k0 + g);
                    vf = *(const v4f*)(Fw + (size_t)yg * D_ + k0 + g);
                }
                __bf16* du = &ldsU[row * PITCH + g];
                __bf16* df = &ldsF[row * PITCH + g];
                du[0]=f2bf(vu.x); du[1]=f2bf(vu.y); du[2]=f2bf(vu.z); du[3]=f2bf(vu.w);
                df[0]=f2bf(vf.x); df[1]=f2bf(vf.y); df[2]=f2bf(vf.z); df[3]=f2bf(vf.w);
            }
            __syncthreads();

            #pragma unroll
            for (int ks = 0; ks < KC; ks += 32) {
                const int arow = w * 16 + l16;
                const int aoff = ks + (half ? 8 : 0);
                v8bf alo = *(const v8bf*)&ldsU[arow * PITCH + aoff];
                v8bf ahi = *(const v8bf*)&ldsU[arow * PITCH + aoff + 16];
                v8bf flo = *(const v8bf*)&ldsF[arow * PITCH + aoff];
                v8bf fhi = *(const v8bf*)&ldsF[arow * PITCH + aoff + 16];
                v16bf aU, aF;
                #pragma unroll
                for (int i = 0; i < 8; ++i) {
                    aU[i] = alo[i]; aU[8+i] = ahi[i];
                    aF[i] = flo[i]; aF[8+i] = fhi[i];
                }
                const int boff = ks + (half ? 8 : 0);
                #pragma unroll
                for (int lt = 0; lt < 8; ++lt) {
                    const int brow = lt * 16 + l16;
                    v8bf blo = *(const v8bf*)&ldsX[brow * PITCH + boff];
                    v8bf bhi = *(const v8bf*)&ldsX[brow * PITCH + boff + 16];
                    v16bf bb;
                    #pragma unroll
                    for (int i = 0; i < 8; ++i) { bb[i] = blo[i]; bb[8+i] = bhi[i]; }
                    accS[lt] = __builtin_amdgcn_wmma_f32_16x16x32_bf16(
                        false, aU, false, bb, (short)0, accS[lt], false, false);
                    accF[lt] = __builtin_amdgcn_wmma_f32_16x16x32_bf16(
                        false, aF, false, bb, (short)0, accF[lt], false, false);
                }
            }
        }

        #pragma unroll
        for (int r = 0; r < 8; ++r) {
            float mx = -1e30f;
            #pragma unroll
            for (int t = 0; t < 8; ++t) mx = fmaxf(mx, accS[t][r]);
            #pragma unroll
            for (int dlt = 1; dlt < 16; dlt <<= 1)
                mx = fmaxf(mx, __shfl_xor(mx, dlt, 32));
            float nm   = fmaxf(run_m[r], mx);
            float corr = __expf(run_m[r] - nm);
            float ps = 0.f, pa = 0.f;
            #pragma unroll
            for (int t = 0; t < 8; ++t) {
                float e = __expf(accS[t][r] - nm);
                ps += e;
                pa += e * accF[t][r];
            }
            #pragma unroll
            for (int dlt = 1; dlt < 16; dlt <<= 1) {
                ps += __shfl_xor(ps, dlt, 32);
                pa += __shfl_xor(pa, dlt, 32);
            }
            run_s[r] = run_s[r] * corr + ps;
            run_a[r] = run_a[r] * corr + pa;
            run_m[r] = nm;
        }
    }

    if (l16 == 0) {
        #pragma unroll
        for (int r = 0; r < 8; ++r) {
            int yg = y0 + w * 16 + r + half * 8;
            if (yg < Y_)
                yout[(size_t)b * Y_ + yg] = run_a[r] / run_s[r] + bias[yg];
        }
    }
}

// Stable BCEWithLogits mean: max(y,0) - t*y + log1p(exp(-|y|))
__global__ __launch_bounds__(256)
void bce_loss_kernel(const float* __restrict__ yv,
                     const float* __restrict__ tgt,
                     float* __restrict__ out) {
    __shared__ float red[256];
    const int N = B_ * Y_;
    float s = 0.f;
    for (int i = threadIdx.x; i < N; i += 256) {
        float yy = yv[i];
        float t  = tgt[i];
        s += fmaxf(yy, 0.f) - yy * t + log1pf(__expf(-fabsf(yy)));
    }
    red[threadIdx.x] = s;
    __syncthreads();
    for (int st = 128; st > 0; st >>= 1) {
        if (threadIdx.x < st) red[threadIdx.x] += red[threadIdx.x + st];
        __syncthreads();
    }
    if (threadIdx.x == 0) out[N] = red[0] / (float)N;
}

extern "C" void kernel_launch(void* const* d_in, const int* in_sizes, int n_in,
                              void* d_out, int out_size, void* d_ws, size_t ws_size,
                              hipStream_t stream) {
    const float* x    = (const float*)d_in[0];  // [B,L,D]
    const float* tgt  = (const float*)d_in[1];  // [B,Y]
    // d_in[2] = text_inputs (int64), unused by the computation
    const float* Uw   = (const float*)d_in[3];  // [Y,D]
    const float* Fw   = (const float*)d_in[4];  // [Y,D]
    const float* bias = (const float*)d_in[5];  // [Y]
    float* out = (float*)d_out;                 // y [B*Y] then loss [1]

    dim3 grid((Y_ + YBLK - 1) / YBLK, B_);      // (70, 16)

    const size_t nX = (size_t)B_ * L_ * D_;     // 16,777,216
    const size_t nW = (size_t)Y_ * D_;          // 4,567,552
    const size_t ws_need = (nX + 2 * nW) * sizeof(__bf16);

    if (ws_size >= ws_need) {
        __bf16* xbf = (__bf16*)d_ws;
        __bf16* ubf = xbf + nX;
        __bf16* fbf = ubf + nW;
        f32_to_bf16_kernel<<<2048, 256, 0, stream>>>(x,  xbf, (int)(nX / 4));
        f32_to_bf16_kernel<<<1024, 256, 0, stream>>>(Uw, ubf, (int)(nW / 4));
        f32_to_bf16_kernel<<<1024, 256, 0, stream>>>(Fw, fbf, (int)(nW / 4));
        const size_t shmem = 2u * 3u * TILE * sizeof(__bf16);   // 110,592 B
        attn_pool_wmma_tdm_kernel<<<grid, 256, shmem, stream>>>(xbf, ubf, fbf,
                                                                bias, out);
    } else {
        attn_pool_wmma_fallback_kernel<<<grid, 256, 0, stream>>>(x, Uw, Fw,
                                                                 bias, out);
    }
    bce_loss_kernel<<<1, 256, 0, stream>>>(out, tgt, out);
}
